// Qwen3Attention_28441273434859
// MI455X (gfx1250) — matrix operand order repaired and checked by reading, not compiled, not run on hardware
//
#include <hip/hip_runtime.h>

// ---------------- CDNA5 WMMA types ----------------
typedef __bf16 v16bf __attribute__((ext_vector_type(16)));
typedef float  v8f   __attribute__((ext_vector_type(8)));

struct FragBF { union { v16bf v; uint4 u[2]; }; };

__device__ __forceinline__ v8f wmma_bf16(const FragBF& a, const FragBF& b, v8f c) {
    return __builtin_amdgcn_wmma_f32_16x16x32_bf16(false, a.v, false, b.v,
                                                   (short)0, c, false, false);
}

// A-fragment (16x32 bf16): lane m=lane&15 holds row m,
// contiguous K = 16*half .. 16*half+15
__device__ __forceinline__ FragBF load_frag_a(const __bf16* base, size_t ld, int k0, int lane) {
    FragBF f;
    const int m = lane & 15, hf = lane >> 4;
    const __bf16* p = base + (size_t)m * ld + k0 + 16 * hf;
    f.u[0] = *(const uint4*)p;
    f.u[1] = *(const uint4*)(p + 8);
    return f;
}

// B-fragment (32x16 bf16) from B^T storage [N][K]: lane n=lane&15 holds column n,
// contiguous K = 16*half .. 16*half+15
__device__ __forceinline__ FragBF load_frag_bt(const __bf16* bt, size_t ld, int k0, int lane) {
    FragBF f;
    const int nn = lane & 15, hf = lane >> 4;
    const __bf16* p = bt + (size_t)nn * ld + k0 + 16 * hf;
    f.u[0] = *(const uint4*)p;
    f.u[1] = *(const uint4*)(p + 8);
    return f;
}

// ---------------- constants ----------------
#define HIDDEN   2048
#define N_HEADS  16
#define N_KV     8
#define HEAD_DIM 128
#define T_TOK    2048
#define SEQ_LEN  1024
#define QKV_N    4096            // 2048 q | 1024 k | 1024 v
#define CACHE_ELEMS (2u * 2048u * 8u * 128u)   // 4,194,304 f32
#define SOFTMAX_SCALE 0.08838834764831845f     // 1/sqrt(128)
#define NEG_BIG (-3.0e38f)

// ---------------- stage 0: conversions ----------------
__global__ void convert_f32_to_bf16(const float* __restrict__ src,
                                    __bf16* __restrict__ dst, int nelem) {
    int i = blockIdx.x * blockDim.x + threadIdx.x;
    if (i < nelem) dst[i] = (__bf16)src[i];
}

// dst[n*K + k] = (bf16) src[k*N + n]   (src is K x N row-major)
__global__ void transpose_convert(const float* __restrict__ src,
                                  __bf16* __restrict__ dst, int K, int N) {
    __shared__ float tile[32][33];
    const int kb = blockIdx.y * 32, nb = blockIdx.x * 32;
    const int tx = threadIdx.x, ty = threadIdx.y;           // 32 x 8
    #pragma unroll
    for (int i = 0; i < 32; i += 8)
        tile[ty + i][tx] = src[(size_t)(kb + ty + i) * N + nb + tx];
    __syncthreads();
    #pragma unroll
    for (int i = 0; i < 32; i += 8)
        dst[(size_t)(nb + ty + i) * K + kb + tx] = (__bf16)tile[tx][ty + i];
}

// ---------------- WMMA GEMM with async global->LDS double buffering ----------
// C[M,N] = A[M,K] * B^T[N,K].  Block = 128 threads (4 waves).
// Block tile: 32(M) x 128(N), K-stage = 64.  Wave computes 16x64 (4 N-tiles).
#define GEMM_BM 32
#define GEMM_BN 128
#define GEMM_BK 64

__global__ __launch_bounds__(128)
void gemm_bf16_wmma(const __bf16* __restrict__ A,
                    const __bf16* __restrict__ BT,
                    float* __restrict__ C,
                    int K, int ldc) {
    __shared__ __align__(16) __bf16 sA[2][GEMM_BM * GEMM_BK];   // 2 x 4KB
    __shared__ __align__(16) __bf16 sB[2][GEMM_BN * GEMM_BK];   // 2 x 16KB
    const int tid  = threadIdx.x;
    const int lane = tid & 31;
    const int wave = tid >> 5;
    const int row0 = blockIdx.x * GEMM_BM;
    const int col0 = blockIdx.y * GEMM_BN;
    const int mhalf = wave & 1;         // 16-row half of the block tile
    const int nquad = wave >> 1;        // 64-col half of the block tile
    const int nn = lane & 15, hf = lane >> 4;

    const __bf16* Ab = A  + (size_t)row0 * K;
    const __bf16* Bb = BT + (size_t)col0 * K;
    const int NS = K / GEMM_BK;

    // One stage = 20KB: A 256 16B-chunks (2/thread) + B 1024 chunks (8/thread)
    // => 10 async-LDS instructions per wave per stage (ASYNCcnt += 10).
    auto issue_stage = [&](int s, int buf) {
        const int k0 = s * GEMM_BK;
        #pragma unroll
        for (int i = 0; i < 2; ++i) {
            const int c = i * 128 + tid;
            const int r = c >> 3, cc = (c & 7) * 8;
            const uint32_t ldsa = (uint32_t)(uintptr_t)&sA[buf][r * GEMM_BK + cc];
            const __bf16* g = Ab + (size_t)r * K + k0 + cc;
            asm volatile("global_load_async_to_lds_b128 %0, %1, off"
                         :: "v"(ldsa), "v"(g) : "memory");
        }
        #pragma unroll
        for (int i = 0; i < 8; ++i) {
            const int c = i * 128 + tid;
            const int r = c >> 3, cc = (c & 7) * 8;
            const uint32_t ldsa = (uint32_t)(uintptr_t)&sB[buf][r * GEMM_BK + cc];
            const __bf16* g = Bb + (size_t)r * K + k0 + cc;
            asm volatile("global_load_async_to_lds_b128 %0, %1, off"
                         :: "v"(ldsa), "v"(g) : "memory");
        }
    };

    v8f acc[4] = {};
    issue_stage(0, 0);
    for (int s = 0; s < NS; ++s) {
        const int buf = s & 1;
        if (s + 1 < NS) {
            issue_stage(s + 1, buf ^ 1);          // overlap next stage with compute
            // async loads complete in order: <=10 outstanding => stage s landed
            asm volatile("s_wait_asynccnt 10" ::: "memory");
        } else {
            asm volatile("s_wait_asynccnt 0" ::: "memory");
        }
        __syncthreads();
        #pragma unroll
        for (int ks = 0; ks < 2; ++ks) {
            const int k0 = ks * 32;
            FragBF a = load_frag_a(&sA[buf][mhalf * 16 * GEMM_BK], GEMM_BK, k0, lane);
            #pragma unroll
            for (int j = 0; j < 4; ++j) {
                FragBF b = load_frag_bt(&sB[buf][(nquad * 64 + j * 16) * GEMM_BK],
                                        GEMM_BK, k0, lane);
                acc[j] = wmma_bf16(a, b, acc[j]);
            }
        }
        __syncthreads();                          // protect buffer before next reuse
    }

    #pragma unroll
    for (int j = 0; j < 4; ++j)
        #pragma unroll
        for (int r = 0; r < 8; ++r)
            C[(size_t)(row0 + mhalf * 16 + r + 8 * hf) * ldc +
              col0 + nquad * 64 + j * 16 + nn] = acc[j][r];
}

// ---------------- stage 2: rmsnorm + rope + cache scatter ----------------
__device__ __forceinline__ float block_sum128(float v, float* red) {
    #pragma unroll
    for (int m = 1; m < 32; m <<= 1) v += __shfl_xor(v, m, 32);
    const int w = threadIdx.x >> 5;
    if ((threadIdx.x & 31) == 0) red[w] = v;
    __syncthreads();
    return red[0] + red[1] + red[2] + red[3];
}

__device__ __forceinline__ float rope_apply(float y, int d, float p) {
    const int i = d >> 1;
    const float inv = __powf(1.0e6f, -(float)(2 * i) * (1.0f / 128.0f));
    const float f = p * inv;
    const float c = __cosf(f), s = __sinf(f);
    const float other = __shfl_xor(y, 1, 32);
    return (d & 1) ? (other * s + y * c) : (y * c - other * s);
}

__global__ void qkv_epilogue(const float* __restrict__ qkv,
                             const float* __restrict__ q_scale,
                             const float* __restrict__ k_scale,
                             const int* __restrict__ pos,
                             const int* __restrict__ slots,
                             float* __restrict__ cache,        // [2][2048][8][128]
                             __bf16* __restrict__ qo,          // [T][16*128]
                             __bf16* __restrict__ ko,          // [T][8*128]
                             __bf16* __restrict__ vTo) {       // [8*128][T]
    __shared__ float red[4];
    const int t = blockIdx.x;
    const int hid = blockIdx.y;                 // 0..15 q heads, 16..23 kv heads
    const int d = threadIdx.x;                  // 128
    const float p = (float)pos[t];

    if (hid < N_HEADS) {
        float x = qkv[(size_t)t * QKV_N + hid * HEAD_DIM + d];
        float ms = block_sum128(x * x, red) * (1.0f / 128.0f);
        float y = x * rsqrtf(ms + 1e-6f) * q_scale[d];
        y = rope_apply(y, d, p);
        qo[(size_t)t * HIDDEN + hid * HEAD_DIM + d] = (__bf16)y;
    } else {
        const int kh = hid - N_HEADS;
        float xk = qkv[(size_t)t * QKV_N + 2048 + kh * HEAD_DIM + d];
        float xv = qkv[(size_t)t * QKV_N + 3072 + kh * HEAD_DIM + d];
        float ms = block_sum128(xk * xk, red) * (1.0f / 128.0f);
        float yk = xk * rsqrtf(ms + 1e-6f) * k_scale[d];
        yk = rope_apply(yk, d, p);
        const int slot = slots[t];
        if (slot >= 0) {
            cache[((size_t)slot * N_KV + kh) * HEAD_DIM + d] = yk;
            cache[(size_t)2048 * N_KV * HEAD_DIM +
                  ((size_t)slot * N_KV + kh) * HEAD_DIM + d] = xv;
        }
        ko[(size_t)t * (N_KV * HEAD_DIM) + kh * HEAD_DIM + d] = (__bf16)yk;
        vTo[((size_t)kh * HEAD_DIM + d) * T_TOK + t] = (__bf16)xv;
    }
}

// ---------------- stage 3: flash attention (wave = 16 q rows) ----------------
__global__ void flash_attn_wmma(const __bf16* __restrict__ qb,   // [T][16*128]
                                const __bf16* __restrict__ kb,   // [T][8*128]
                                const __bf16* __restrict__ vT,   // [8*128][T]
                                __bf16* __restrict__ attn_out) { // [T][16*128]
    __shared__ __align__(16) __bf16 ldsP[4][16][32];
    const int lane = threadIdx.x & 31;
    const int wave = threadIdx.x >> 5;
    const int h    = blockIdx.x;        // 16 heads
    const int qblk = blockIdx.y;        // 16 blocks of 64 rows
    const int seq  = blockIdx.z;        // 2 sequences
    const int kvh  = h >> 1;            // GQA: 2 q heads per kv head
    const int ln = lane & 15, hf = lane >> 4;

    const int qloc  = qblk * 64 + wave * 16;       // local (in-seq) row base
    const int qglob = seq * SEQ_LEN + qloc;

    FragBF qf[4];
    #pragma unroll
    for (int kk = 0; kk < 4; ++kk)
        qf[kk] = load_frag_a(qb + (size_t)qglob * HIDDEN + h * HEAD_DIM,
                             HIDDEN, kk * 32, lane);

    v8f acc[8] = {};
    float mi[8], li[8];
    #pragma unroll
    for (int r = 0; r < 8; ++r) { mi[r] = NEG_BIG; li[r] = 0.0f; }

    const int kmax = qloc + 15;
    for (int kbase = 0; kbase <= kmax; kbase += 32) {
        if (kbase + 32 <= kmax)   // prefetch next key block
            __builtin_prefetch(kb + (size_t)(seq * SEQ_LEN + kbase + 32 + ln) *
                                      (N_KV * HEAD_DIM) + kvh * HEAD_DIM, 0, 1);
        // ---- scores: two 16x16 tiles over 32 keys ----
        v8f s0 = {}, s1 = {};
        #pragma unroll
        for (int kk = 0; kk < 4; ++kk) {
            FragBF b0, b1;
            const __bf16* p = kb + (size_t)(seq * SEQ_LEN + kbase + ln) * (N_KV * HEAD_DIM)
                                 + kvh * HEAD_DIM + kk * 32 + 16 * hf;
            b0.u[0] = *(const uint4*)p;
            b0.u[1] = *(const uint4*)(p + 8);
            const __bf16* p1 = p + (size_t)16 * (N_KV * HEAD_DIM);
            b1.u[0] = *(const uint4*)p1;
            b1.u[1] = *(const uint4*)(p1 + 8);
            s0 = wmma_bf16(qf[kk], b0, s0);
            s1 = wmma_bf16(qf[kk], b1, s1);
        }

        // ---- online softmax (per-lane rows M = r + 8*hf) ----
        float p0[8], p1v[8];
        #pragma unroll
        for (int r = 0; r < 8; ++r) {
            const int qp = qloc + r + 8 * hf;
            float v0 = s0[r] * SOFTMAX_SCALE;
            float v1 = s1[r] * SOFTMAX_SCALE;
            if (kbase + ln > qp)      v0 = NEG_BIG;
            if (kbase + 16 + ln > qp) v1 = NEG_BIG;
            float mx = fmaxf(v0, v1);
            #pragma unroll
            for (int m = 1; m <= 8; m <<= 1) mx = fmaxf(mx, __shfl_xor(mx, m, 32));
            const float nm = fmaxf(mi[r], mx);
            p0[r]  = __expf(v0 - nm);
            p1v[r] = __expf(v1 - nm);
            float sum = p0[r] + p1v[r];
            #pragma unroll
            for (int m = 1; m <= 8; m <<= 1) sum += __shfl_xor(sum, m, 32);
            const float alpha = __expf(mi[r] - nm);
            li[r] = li[r] * alpha + sum;
            mi[r] = nm;
            #pragma unroll
            for (int dt = 0; dt < 8; ++dt) acc[dt][r] *= alpha;
        }

        // ---- probs -> LDS (D layout -> row-major), then reload as A-frag ----
        #pragma unroll
        for (int r = 0; r < 8; ++r) {
            ldsP[wave][r + 8 * hf][ln]      = (__bf16)p0[r];
            ldsP[wave][r + 8 * hf][16 + ln] = (__bf16)p1v[r];
        }
        asm volatile("s_wait_dscnt 0" ::: "memory");   // wave-local LDS ordering
        FragBF pa;
        pa.u[0] = *(const uint4*)&ldsP[wave][ln][16 * hf];
        pa.u[1] = *(const uint4*)&ldsP[wave][ln][16 * hf + 8];

        // ---- P @ V : 8 head-dim tiles, K = 32 tokens ----
        #pragma unroll
        for (int dt = 0; dt < 8; ++dt) {
            FragBF bv;
            const __bf16* p = vT + ((size_t)kvh * HEAD_DIM + dt * 16 + ln) * T_TOK
                                 + seq * SEQ_LEN + kbase + 16 * hf;
            bv.u[0] = *(const uint4*)p;
            bv.u[1] = *(const uint4*)(p + 8);
            acc[dt] = wmma_bf16(pa, bv, acc[dt]);
        }
    }

    // ---- normalize + store bf16 attn for the output projection ----
    #pragma unroll
    for (int dt = 0; dt < 8; ++dt)
        #pragma unroll
        for (int r = 0; r < 8; ++r) {
            const float o = acc[dt][r] / li[r];
            attn_out[(size_t)(qglob + r + 8 * hf) * HIDDEN +
                     h * HEAD_DIM + dt * 16 + ln] = (__bf16)o;
        }
}

// ---------------- launcher ----------------
extern "C" void kernel_launch(void* const* d_in, const int* in_sizes, int n_in,
                              void* d_out, int out_size, void* d_ws, size_t ws_size,
                              hipStream_t stream) {
    const float* x    = (const float*)d_in[0];
    const float* kvc  = (const float*)d_in[1];
    const float* wq   = (const float*)d_in[2];
    const float* qsc  = (const float*)d_in[3];
    const float* wk   = (const float*)d_in[4];
    const float* ksc  = (const float*)d_in[5];
    const float* wv   = (const float*)d_in[6];
    const float* wo   = (const float*)d_in[7];
    const int*   pos  = (const int*)d_in[8];
    const int*   slot = (const int*)d_in[9];

    float* out   = (float*)d_out;
    float* cache = out;                       // 2*2048*8*128 f32
    float* obuf  = out + CACHE_ELEMS;         // 2048*2048 f32

    char* ws = (char*)d_ws;
    __bf16* wT   = (__bf16*)(ws);               // [4096][2048] bf16 (q|k|v B^T)
    __bf16* woT  = (__bf16*)(ws + 16777216);    // [2048][2048] bf16
    __bf16* xbf  = (__bf16*)(ws + 25165824);    // [2048][2048] bf16; later attn_bf16
    float*  qkv  = (float*) (ws + 33554432);    // [2048][4096] f32
    __bf16* qbf  = (__bf16*)(ws + 67108864);    // [2048][2048] bf16
    __bf16* kbf  = (__bf16*)(ws + 75497472);    // [2048][1024] bf16
    __bf16* vTbf = (__bf16*)(ws + 79691776);    // [1024][2048] bf16

    // kv cache passthrough (all slots are rewritten here, but keep ref semantics)
    hipMemcpyAsync(cache, kvc, (size_t)CACHE_ELEMS * sizeof(float),
                   hipMemcpyDeviceToDevice, stream);

    // stage 0: conversions / transposes
    convert_f32_to_bf16<<<(T_TOK * HIDDEN) / 256, 256, 0, stream>>>(x, xbf, T_TOK * HIDDEN);
    {
        dim3 blk(32, 8);
        transpose_convert<<<dim3(2048 / 32, 2048 / 32), blk, 0, stream>>>(wq, wT, 2048, 2048);
        transpose_convert<<<dim3(1024 / 32, 2048 / 32), blk, 0, stream>>>(wk, wT + (size_t)2048 * 2048, 2048, 1024);
        transpose_convert<<<dim3(1024 / 32, 2048 / 32), blk, 0, stream>>>(wv, wT + (size_t)3072 * 2048, 2048, 1024);
        transpose_convert<<<dim3(2048 / 32, 2048 / 32), blk, 0, stream>>>(wo, woT, 2048, 2048);
    }

    // stage 1: QKV projection  qkv[2048][4096] = xbf * wT^T
    gemm_bf16_wmma<<<dim3(T_TOK / GEMM_BM, QKV_N / GEMM_BN), 128, 0, stream>>>(
        xbf, wT, qkv, HIDDEN, QKV_N);

    // stage 2: rmsnorm + rope + cache scatter + bf16 operand prep
    qkv_epilogue<<<dim3(T_TOK, N_HEADS + N_KV), HEAD_DIM, 0, stream>>>(
        qkv, qsc, ksc, pos, slot, cache, qbf, kbf, vTbf);

    // stage 3: attention (attn output reuses the xbf region as bf16 [2048][2048])
    flash_attn_wmma<<<dim3(N_HEADS, SEQ_LEN / 64, T_TOK / SEQ_LEN), 128, 0, stream>>>(
        qbf, kbf, vTbf, xbf);

    // stage 4: output projection  o[2048][2048] = attn_bf16 * woT^T
    gemm_bf16_wmma<<<dim3(T_TOK / GEMM_BM, HIDDEN / GEMM_BN), 128, 0, stream>>>(
        xbf, woT, obuf, HIDDEN, HIDDEN);
}
